// ST_ATT_Layer_78889959293647
// MI455X (gfx1250) — matrix operand order repaired and checked by reading, not compiled, not run on hardware
//
#include <hip/hip_runtime.h>
#include <hip/hip_bf16.h>

#if defined(__has_builtin)
#if __has_builtin(__builtin_amdgcn_global_load_async_to_lds_b128) && \
    __has_builtin(__builtin_amdgcn_s_wait_asynccnt)
#define HAVE_ASYNC 1
#endif
#endif
#ifndef HAVE_ASYNC
#define HAVE_ASYNC 0
#endif

typedef __attribute__((ext_vector_type(16))) _Float16 v16h;
typedef __attribute__((ext_vector_type(8)))  float    v8f;
typedef __attribute__((ext_vector_type(4)))  int      v4i;

#define BATCH  256
#define SEQN   176
#define DMODEL 512
#define HEADS  8
#define DHEAD  64
#define ROWS   (BATCH*SEQN)   // 45056

// ---------------------------------------------------------------------------
// async 16B global -> LDS copy (ASYNCcnt path); no-op stub if unavailable
// builtin signature (probe-confirmed): (v4i AS1* src, v4i AS3* dst, imm, imm)
// ---------------------------------------------------------------------------
__device__ __forceinline__ void async_cp16(const _Float16* g, _Float16* l) {
#if HAVE_ASYNC
    __builtin_amdgcn_global_load_async_to_lds_b128(
        (__attribute__((address_space(1))) v4i*)g,
        (__attribute__((address_space(3))) v4i*)l, 0, 0);
#else
    (void)g; (void)l;
#endif
}

__device__ __forceinline__ void async_wait0() {
#if HAVE_ASYNC
    __builtin_amdgcn_s_wait_asynccnt(0);
#endif
}

// ---------------------------------------------------------------------------
// Kernel 1: x + positional encoding -> f16
// ---------------------------------------------------------------------------
__global__ __launch_bounds__(256) void prep_pe_f16(const float* __restrict__ x,
                                                   _Float16* __restrict__ xh) {
    int idx = blockIdx.x * 256 + threadIdx.x;
    if (idx >= ROWS * DMODEL) return;
    int d = idx & (DMODEL - 1);
    int s = (idx >> 9) % SEQN;
    float de   = (float)(d & ~1);
    float freq = __expf(de * (-9.210340371976184f / 512.0f)); // -ln(10000)/512
    float ang  = (float)s * freq;
    float pe   = (d & 1) ? __cosf(ang) : __sinf(ang);
    xh[idx] = (_Float16)(x[idx] + pe);
}

// ---------------------------------------------------------------------------
// Kernel 2: convert the four [512,512] weights to f16 (packed contiguously)
// ---------------------------------------------------------------------------
__global__ __launch_bounds__(256) void wcvt(const float* __restrict__ wq,
                                            const float* __restrict__ wk,
                                            const float* __restrict__ wv,
                                            const float* __restrict__ wo,
                                            _Float16* __restrict__ wh) {
    int idx = blockIdx.x * 256 + threadIdx.x;
    if (idx >= 4 * DMODEL * DMODEL) return;
    int which = idx >> 18;          // 262144 elems per weight
    int off   = idx & 262143;
    const float* w = which == 0 ? wq : which == 1 ? wk : which == 2 ? wv : wo;
    wh[idx] = (_Float16)w[off];
}

// ---------------------------------------------------------------------------
// Kernel 3/5: Y = act(X @ W^T + bias).  X:[45056,512] f16, W:[512,512] f16
// MODE 0: f16 output scattered to [B,H,S,Dh]  (blockIdx.y == head)
// MODE 1: f32 output row-major [45056,512]
// Block = 256 thr = 8 waves; tile 128(M) x 64(N), K stepped by 32.
// With async: double-buffered LDS tiles, tile k+1 streams in during WMMAs on k.
// ---------------------------------------------------------------------------
template <int MODE>
__global__ __launch_bounds__(256) void gemm_wmma(const _Float16* __restrict__ X,
                                                 const _Float16* __restrict__ W,
                                                 const float* __restrict__ bias,
                                                 _Float16* __restrict__ outh,
                                                 float* __restrict__ outf,
                                                 int relu) {
#if HAVE_ASYNC
    __shared__ _Float16 As[2][128 * 40];   // 2 x 10 KB
    __shared__ _Float16 Bs[2][64 * 40];    // 2 x 5 KB
#else
    __shared__ _Float16 As[1][128 * 40];
    __shared__ _Float16 Bs[1][64 * 40];
#endif

    const int tid  = threadIdx.x;
    const int wave = tid >> 5;
    const int lane = tid & 31;
    const int mBase = blockIdx.x * 128;
    const int nBase = blockIdx.y * 64;

    const int laneM = lane & 15;
    const int kbA = (lane < 16) ? 0 : 16;   // A frag K-split (contiguous, matches B)
    const int kbB = (lane < 16) ? 0 : 16;   // B frag K-split
    const int m0  = wave * 16;

    v8f acc[4] = {};

    auto stage = [&](int buf, int k0) {
#if HAVE_ASYNC
        // A tile: 128x32 halves = 512 x 16B chunks (2/thread), B: 256 chunks
#pragma unroll
        for (int j = 0; j < 2; ++j) {
            int c = tid + j * 256;
            int r = c >> 2, cc = (c & 3) * 8;
            async_cp16(&X[(size_t)(mBase + r) * DMODEL + k0 + cc], &As[buf][r * 40 + cc]);
        }
        {
            int r = tid >> 2, cc = (tid & 3) * 8;
            async_cp16(&W[(size_t)(nBase + r) * DMODEL + k0 + cc], &Bs[buf][r * 40 + cc]);
        }
#else
#pragma unroll
        for (int j = 0; j < 4; ++j) {
            int c = tid + j * 256;
            int r = c >> 3, cc = (c & 7) * 4;
            *(unsigned long long*)&As[buf][r * 40 + cc] =
                *(const unsigned long long*)&X[(size_t)(mBase + r) * DMODEL + k0 + cc];
        }
#pragma unroll
        for (int j = 0; j < 2; ++j) {
            int c = tid + j * 256;
            int r = c >> 3, cc = (c & 7) * 4;
            *(unsigned long long*)&Bs[buf][r * 40 + cc] =
                *(const unsigned long long*)&W[(size_t)(nBase + r) * DMODEL + k0 + cc];
        }
#endif
    };

#if HAVE_ASYNC
    stage(0, 0);
#endif
    for (int kk = 0; kk < 16; ++kk) {
#if HAVE_ASYNC
        async_wait0();          // tile kk resident (own wave's transfers)
        __syncthreads();        // ... and everyone else's
        if (kk < 15) stage((kk + 1) & 1, (kk + 1) * 32);
        const int buf = kk & 1;
#else
        stage(0, kk * 32);
        __syncthreads();
        const int buf = 0;
#endif
        v16h a;
        const int rowA = (m0 + laneM) * 40;
#pragma unroll
        for (int i = 0; i < 16; ++i) {
            a[i] = As[buf][rowA + kbA + i];
        }
#pragma unroll
        for (int n = 0; n < 4; ++n) {
            v16h b;
            const int rowB = (n * 16 + laneM) * 40 + kbB;
#pragma unroll
            for (int e = 0; e < 16; ++e) b[e] = Bs[buf][rowB + e];
            acc[n] = __builtin_amdgcn_wmma_f32_16x16x32_f16(
                false, a, false, b, (short)0, acc[n], false, false);
        }
#if !HAVE_ASYNC
        __syncthreads();
#endif
    }

    // epilogue
    const int rHi = (lane < 16) ? 0 : 8;
#pragma unroll
    for (int n = 0; n < 4; ++n) {
        const int col = nBase + n * 16 + laneM;
        const float bv = bias[col];
#pragma unroll
        for (int r = 0; r < 8; ++r) {
            const int row = mBase + m0 + r + rHi;
            float v = acc[n][r] + bv;
            if (relu) v = v > 0.f ? v : 0.f;
            if (MODE == 0) {
                int b = row / SEQN, s = row % SEQN;
                int h = col >> 6,  d = col & 63;
                outh[(((size_t)(b * HEADS + h)) * SEQN + s) * DHEAD + d] = (_Float16)v;
            } else {
                outf[(size_t)row * DMODEL + col] = v;
            }
        }
    }
}

// ---------------------------------------------------------------------------
// Kernel 4: one block per (b,h): scores -> mask -> softmax -> attn@V
// All of Q, K, V^T and padded scores live in LDS (~134 KB).
// ---------------------------------------------------------------------------
__global__ __launch_bounds__(256) void attn_wmma(const _Float16* __restrict__ Q,
                                                 const _Float16* __restrict__ K,
                                                 const _Float16* __restrict__ V,
                                                 _Float16* __restrict__ ctx) {
    __shared__ _Float16 Qs[SEQN * DHEAD];    // 22528 B
    __shared__ _Float16 Ks[SEQN * DHEAD];    // 22528 B
    __shared__ _Float16 VsT[DHEAD * 192];    // V transposed, K padded to 192
    __shared__ _Float16 Ss[SEQN * 192];      // scores / attn, cols padded

    const int tid  = threadIdx.x;
    const int wave = tid >> 5;
    const int lane = tid & 31;
    const int bh   = blockIdx.x;
    const size_t base = (size_t)bh * SEQN * DHEAD;

#if HAVE_ASYNC
    // Q,K: 176*64 halves = 1408 x 16B chunks each, streamed via ASYNCcnt
#pragma unroll
    for (int j = 0; j < 6; ++j) {
        int c = tid + j * 256;
        if (c < 1408) {
            async_cp16(&Q[base + c * 8], &Qs[c * 8]);
            async_cp16(&K[base + c * 8], &Ks[c * 8]);
        }
    }
#else
#pragma unroll
    for (int j = 0; j < 11; ++j) {
        int c = (tid + j * 256) * 4;
        *(unsigned long long*)&Qs[c] = *(const unsigned long long*)&Q[base + c];
        *(unsigned long long*)&Ks[c] = *(const unsigned long long*)&K[base + c];
    }
#endif
    // V transposed into LDS (per-element; transpose precludes async path)
    for (int j = 0; j < 44; ++j) {
        int e = tid + j * 256;
        int k = e >> 6, d = e & 63;
        VsT[d * 192 + k] = V[base + e];
    }
    // zero pads: VsT k=176..191, Ss cols 176..191
#pragma unroll
    for (int j = 0; j < 4; ++j) {
        int e = tid + j * 256;                 // 64*16
        VsT[(e >> 4) * 192 + 176 + (e & 15)] = (_Float16)0.f;
    }
#pragma unroll
    for (int j = 0; j < 11; ++j) {
        int e = tid + j * 256;                 // 176*16
        Ss[(e >> 4) * 192 + 176 + (e & 15)] = (_Float16)0.f;
    }
    async_wait0();
    __syncthreads();

    const int laneM = lane & 15;
    const int kbA = (lane < 16) ? 0 : 16;
    const int kbB = (lane < 16) ? 0 : 16;
    const int rHi = (lane < 16) ? 0 : 8;

    // ---- scores = (Q K^T)/8, masked ----
    for (int t = wave; t < 121; t += 8) {        // 11x11 tiles of 16x16
        const int ti = t / 11, tj = t % 11;
        v8f acc = {};
#pragma unroll
        for (int kk = 0; kk < 2; ++kk) {
            const int k0 = kk * 32;
            v16h a, b;
            const int rowA = (ti * 16 + laneM) * DHEAD + k0;
#pragma unroll
            for (int i = 0; i < 16; ++i) {
                a[i] = Qs[rowA + kbA + i];
            }
            const int rowB = (tj * 16 + laneM) * DHEAD + k0 + kbB;
#pragma unroll
            for (int e = 0; e < 16; ++e) b[e] = Ks[rowB + e];
            acc = __builtin_amdgcn_wmma_f32_16x16x32_f16(
                false, a, false, b, (short)0, acc, false, false);
        }
#pragma unroll
        for (int r = 0; r < 8; ++r) {
            const int i = ti * 16 + r + rHi;
            const int j = tj * 16 + laneM;
            const float sc = acc[r] * 0.125f;     // 1/sqrt(64)
            const bool masked = (i / 22 == j / 22) && (i != j); // temporal mask
            Ss[i * 192 + j] = masked ? (_Float16)(-__builtin_inff()) : (_Float16)sc;
        }
    }
    __syncthreads();

    // ---- softmax per row (one thread per row; no WMMA under divergence) ----
    if (tid < SEQN) {
        _Float16* row = &Ss[tid * 192];
        float m = -__builtin_inff();
        for (int j = 0; j < SEQN; ++j) { float v = (float)row[j]; m = v > m ? v : m; }
        float sum = 0.f;
        for (int j = 0; j < SEQN; ++j) {
            float e = __expf((float)row[j] - m);
            sum += e;
            row[j] = (_Float16)e;
        }
        const float inv = 1.f / sum;
        for (int j = 0; j < SEQN; ++j) row[j] = (_Float16)((float)row[j] * inv);
    }
    __syncthreads();

    // ---- out = attn @ V : [176,192] @ [192,64] ----
    const int b_ = bh >> 3, h_ = bh & 7;
    for (int t = wave; t < 44; t += 8) {          // 11 M-tiles x 4 N-tiles
        const int ti = t >> 2, n0 = (t & 3) * 16;
        v8f acc = {};
#pragma unroll
        for (int kk = 0; kk < 6; ++kk) {
            const int k0 = kk * 32;
            v16h a, bb;
            const int rowA = (ti * 16 + laneM) * 192 + k0;
#pragma unroll
            for (int i = 0; i < 16; ++i) {
                a[i] = Ss[rowA + kbA + i];
            }
            const int rowB = (n0 + laneM) * 192 + k0 + kbB;
#pragma unroll
            for (int e = 0; e < 16; ++e) bb[e] = VsT[rowB + e];
            acc = __builtin_amdgcn_wmma_f32_16x16x32_f16(
                false, a, false, bb, (short)0, acc, false, false);
        }
#pragma unroll
        for (int r = 0; r < 8; ++r) {
            const int s = ti * 16 + r + rHi;
            const int d = n0 + laneM;
            ctx[((size_t)(b_ * SEQN + s)) * DMODEL + h_ * DHEAD + d] = (_Float16)acc[r];
        }
    }
}

// ---------------------------------------------------------------------------
// Kernel 6: in-place LayerNorm (torch-style: unbiased std, /(std+eps))
// one block per row of 512
// ---------------------------------------------------------------------------
__global__ __launch_bounds__(256) void layernorm_inplace(float* __restrict__ y,
                                                         const float* __restrict__ a2,
                                                         const float* __restrict__ b2) {
    __shared__ float red[256];
    const int tid = threadIdx.x;
    float* row = y + (size_t)blockIdx.x * DMODEL;
    const float v0 = row[tid], v1 = row[tid + 256];
    red[tid] = v0 + v1;
    __syncthreads();
    for (int s = 128; s > 0; s >>= 1) {
        if (tid < s) red[tid] += red[tid + s];
        __syncthreads();
    }
    const float mean = red[0] * (1.0f / 512.0f);
    __syncthreads();
    const float d0 = v0 - mean, d1 = v1 - mean;
    red[tid] = d0 * d0 + d1 * d1;
    __syncthreads();
    for (int s = 128; s > 0; s >>= 1) {
        if (tid < s) red[tid] += red[tid + s];
        __syncthreads();
    }
    const float stdv = sqrtf(red[0] * (1.0f / 511.0f));   // ddof=1
    const float inv  = 1.0f / (stdv + 1e-6f);
    row[tid]       = a2[tid]       * d0 * inv + b2[tid];
    row[tid + 256] = a2[tid + 256] * d1 * inv + b2[tid + 256];
}

// ---------------------------------------------------------------------------
extern "C" void kernel_launch(void* const* d_in, const int* in_sizes, int n_in,
                              void* d_out, int out_size, void* d_ws, size_t ws_size,
                              hipStream_t stream) {
    const float* x  = (const float*)d_in[0];
    const float* wq = (const float*)d_in[1];
    const float* bq = (const float*)d_in[2];
    const float* wk = (const float*)d_in[3];
    const float* bk = (const float*)d_in[4];
    const float* wv = (const float*)d_in[5];
    const float* bv = (const float*)d_in[6];
    const float* wo = (const float*)d_in[7];
    const float* bo = (const float*)d_in[8];
    const float* a2 = (const float*)d_in[9];
    const float* b2 = (const float*)d_in[10];
    float* out = (float*)d_out;

    char* ws = (char*)d_ws;
    const size_t tensH = (size_t)ROWS * DMODEL * sizeof(_Float16);   // 46,137,344 B
    const size_t wH    = (size_t)4 * DMODEL * DMODEL * sizeof(_Float16);
    _Float16* xh  = (_Float16*)ws;                       // x+PE in f16 (reused as ctx)
    _Float16* wh  = (_Float16*)(ws + tensH);
    _Float16* qh  = (_Float16*)(ws + tensH + wH);
    _Float16* kh  = qh + (size_t)ROWS * DMODEL;
    _Float16* vh  = kh + (size_t)ROWS * DMODEL;
    _Float16* wqh = wh;
    _Float16* wkh = wh + DMODEL * DMODEL;
    _Float16* wvh = wh + 2 * DMODEL * DMODEL;
    _Float16* woh = wh + 3 * DMODEL * DMODEL;
    _Float16* ctx = xh;   // alias: xh dead after the QKV GEMMs

    prep_pe_f16<<<(ROWS * DMODEL + 255) / 256, 256, 0, stream>>>(x, xh);
    wcvt<<<(4 * DMODEL * DMODEL + 255) / 256, 256, 0, stream>>>(wq, wk, wv, wo, wh);

    dim3 g(ROWS / 128, DMODEL / 64);
    gemm_wmma<0><<<g, 256, 0, stream>>>(xh, wqh, bq, qh, nullptr, 0);
    gemm_wmma<0><<<g, 256, 0, stream>>>(xh, wkh, bk, kh, nullptr, 0);
    gemm_wmma<0><<<g, 256, 0, stream>>>(xh, wvh, bv, vh, nullptr, 1);  // ReLU(V)

    attn_wmma<<<BATCH * HEADS, 256, 0, stream>>>(qh, kh, vh, ctx);

    gemm_wmma<1><<<g, 256, 0, stream>>>(ctx, woh, bo, nullptr, out, 1); // ReLU

    layernorm_inplace<<<ROWS, 256, 0, stream>>>(out, a2, b2);
}
